// ChargeAwarePotentialClean_2336462209090
// MI455X (gfx1250) — compile-verified, matrix operand order checked by reading, not run on hardware
//
#include <hip/hip_runtime.h>
#include <hip/hip_bf16.h>
#include <stdint.h>

// ---------------------------------------------------------------------------
// ChargeAwarePotential for MI455X (gfx1250, wave32, WMMA).
// Message-passing matmul (N x N x 64) runs on v_wmma_f32_16x16x32_bf16 with
// f32 accumulation; pair weights W = exp(-r/2) are generated in registers
// (never materialized to HBM). All per-atom math stays f32 VALU.
// ---------------------------------------------------------------------------

typedef __attribute__((ext_vector_type(16))) __bf16 v16bf;
typedef __attribute__((ext_vector_type(8)))  float  v8f;
typedef __attribute__((ext_vector_type(8)))  unsigned int v8u;

#define HID 64
#define K_E 14.3996f
#define N_ELEM 119

__device__ __forceinline__ uint32_t pack_bf16x2(float lo, float hi) {
    __bf16 l = (__bf16)lo;
    __bf16 h = (__bf16)hi;
    unsigned short lu = __builtin_bit_cast(unsigned short, l);
    unsigned short hu = __builtin_bit_cast(unsigned short, h);
    return (uint32_t)lu | ((uint32_t)hu << 16);
}

__device__ __forceinline__ float silu_f(float x) {
    return x / (1.0f + __expf(-x));
}

// ---------------------------------------------------------------------------
// K1: h0[i][c] = atom_embed[Z[i]][c] + q_embed[Q+5][c] + s_embed[S-1][c]
// ---------------------------------------------------------------------------
__global__ void embed_kernel(const int* __restrict__ Z,
                             const int* __restrict__ Q,
                             const int* __restrict__ S,
                             const float* __restrict__ atom_embed,
                             const float* __restrict__ q_embed,
                             const float* __restrict__ s_embed,
                             float* __restrict__ h0, int n) {
    int idx = blockIdx.x * blockDim.x + threadIdx.x;
    if (idx >= n * HID) return;
    int i = idx >> 6;
    int c = idx & 63;
    h0[idx] = atom_embed[Z[i] * HID + c]
            + q_embed[(Q[0] + 5) * HID + c]
            + s_embed[(S[0] - 1) * HID + c];
}

// ---------------------------------------------------------------------------
// K2: repack h0 into WMMA B-matrix lane layout, packed bf16x2.
// B tile for j-block jb, hid-tile t:  B[k][nc] = h0[jb*32+k][t*16+nc]
// lane layout (16-bit B, 32x16): lane = {khalf[4], nc[3:0]},
//   VGPR v holds K = (v&4 ? 16:0) + khalf*8 + (v&3)*2 + {0,1}
// Storage: bpack[((jb*4 + t)*32 + lane)*8 + v]  -> per-lane contiguous 32B.
// ---------------------------------------------------------------------------
__global__ void repack_kernel(const float* __restrict__ h0,
                              uint32_t* __restrict__ bpack, int n) {
    int idx = blockIdx.x * blockDim.x + threadIdx.x;
    int total = (n / 32) * 4 * 32 * 8;
    if (idx >= total) return;
    int v     = idx & 7;
    int lane  = (idx >> 3) & 31;
    int t     = (idx >> 8) & 3;
    int jb    = idx >> 10;
    int nc    = lane & 15;
    int khalf = lane >> 4;
    int k0    = ((v & 4) ? 16 : 0) + khalf * 8 + (v & 3) * 2;
    float lo = h0[(jb * 32 + k0    ) * HID + t * 16 + nc];
    float hi = h0[(jb * 32 + k0 + 1) * HID + t * 16 + nc];
    bpack[idx] = pack_bf16x2(lo, hi);
}

// ---------------------------------------------------------------------------
// K3: message passing  h = h0 + W @ h0  via WMMA.
// blockDim = 256 (8 waves); each wave owns a 16-row output block (16 x 64),
// held in 4x v8f accumulators. j-loop in steps of 32 (one WMMA K-slab).
// Loop body order is chosen so the only load-wait lands right before the
// WMMAs, hidden behind the transcendental A-tile build:
//   barrier / LDS-stage positions / barrier  ->  prefetch + 8x b128 B loads
//   ->  A build (v_sqrt + v_exp, ~150 VALU)  ->  4x WMMA.
// Diagonal trick: w(i,i) = exp(-0.5e-6) rounds to exactly 1.0 in bf16, so we
// leave the diagonal unmasked (zero ops in the loop) and subtract the known
// contribution 1.0*bf16(h0[row][col]) once at writeback.
// ---------------------------------------------------------------------------
__global__ void msgpass_kernel(const float* __restrict__ Rp,
                               const float* __restrict__ h0,
                               const uint32_t* __restrict__ bpack,
                               float* __restrict__ hout, int n) {
    const int lane  = threadIdx.x & 31;
    const int wave  = threadIdx.x >> 5;
    const int ib    = blockIdx.x * 8 + wave;       // 16-row block index
    const int m     = lane & 15;
    const int khalf = lane >> 4;
    const int i     = ib * 16 + m;                 // global row for A entries

    const float xi = Rp[3 * i + 0];
    const float yi = Rp[3 * i + 1];
    const float zi = Rp[3 * i + 2];

    __shared__ float sRx[32];
    __shared__ float sRy[32];
    __shared__ float sRz[32];

    v8f acc[4] = {{}, {}, {}, {}};

    const int nblk = n / 32;
    for (int jb = 0; jb < nblk; ++jb) {
        // ---- stage this slab's 32 positions in LDS (block-shared) ----
        __syncthreads();
        if (threadIdx.x < 32) {
            const int j = jb * 32 + threadIdx.x;
            sRx[threadIdx.x] = Rp[3 * j + 0];
            sRy[threadIdx.x] = Rp[3 * j + 1];
            sRz[threadIdx.x] = Rp[3 * j + 2];
        }
        __syncthreads();

        if (jb + 1 < nblk) {
            // pull next 4KB B slab toward the WGP (128B per lane covers it)
            __builtin_prefetch(bpack + (size_t)(jb + 1) * 1024 + lane * 32, 0, 3);
        }

        // ---- issue all B loads; the A build below hides their latency ----
        v8u braw[4];
#pragma unroll
        for (int t = 0; t < 4; ++t) {
            braw[t] = *reinterpret_cast<const v8u*>(
                bpack + ((size_t)jb * 4 + t) * 256 + lane * 8);
        }

        // ---- build A tile: W[m, k] for this lane's 16 K values ----
        v16bf a;
#pragma unroll
        for (int v = 0; v < 8; ++v) {
#pragma unroll
            for (int p = 0; p < 2; ++p) {
                const int k = ((v & 4) ? 16 : 0) + khalf * 8 + (v & 3) * 2 + p;
                const float dx = xi - sRx[k];
                const float dy = yi - sRy[k];
                const float dz = zi - sRz[k];
                const float r2  = fmaxf(dx * dx + dy * dy + dz * dz, 1e-12f);
                const float rij = __builtin_amdgcn_sqrtf(r2);   // raw v_sqrt_f32
                a[v * 2 + p] = (__bf16)__expf(-0.5f * rij);
            }
        }

        // ---- 4 hid tiles on the matrix pipe ----
#pragma unroll
        for (int t = 0; t < 4; ++t) {
            v16bf b = __builtin_bit_cast(v16bf, braw[t]);
            acc[t] = __builtin_amdgcn_wmma_f32_16x16x32_bf16(
                false, a, false, b, (short)0, acc[t], false, false);
        }
    }

    // ---- writeback: D layout: lane -> n = lane&15, VGPR r -> M = r + 8*khalf
    // Remove the (exactly known) diagonal contribution 1.0 * bf16(h0[row][col]).
#pragma unroll
    for (int t = 0; t < 4; ++t) {
#pragma unroll
        for (int r = 0; r < 8; ++r) {
            const int row = ib * 16 + khalf * 8 + r;
            const int col = t * 16 + m;
            const float base = h0[row * HID + col];
            const float diag = (float)(__bf16)base;   // what the WMMA summed
            hout[row * HID + col] = base + (acc[t][r] - diag);
        }
    }
}

// ---------------------------------------------------------------------------
// K4: per-row MLPs (f32).  One block (64 threads) per row.
//   E_row[i]  = silu(h @ eW1 + eb1) @ eW2 + eb2
//   q_raw[i]  = silu(h @ cW1 + cb1) @ cW2 + cb2
// ---------------------------------------------------------------------------
__global__ void row_mlp_kernel(const float* __restrict__ h,
                               const float* __restrict__ eW1, const float* __restrict__ eb1,
                               const float* __restrict__ eW2, const float* __restrict__ eb2,
                               const float* __restrict__ cW1, const float* __restrict__ cb1,
                               const float* __restrict__ cW2, const float* __restrict__ cb2,
                               float* __restrict__ E_row, float* __restrict__ q_raw, int n) {
    const int i = blockIdx.x;
    const int u = threadIdx.x;
    __shared__ float hrow[HID];
    __shared__ float red[HID];

    hrow[u] = h[i * HID + u];
    __syncthreads();

    float se = eb1[u];
    float sc = cb1[u];
#pragma unroll 8
    for (int k = 0; k < HID; ++k) {
        const float hv = hrow[k];
        se = fmaf(hv, eW1[k * HID + u], se);
        sc = fmaf(hv, cW1[k * HID + u], sc);
    }
    const float pe = silu_f(se) * eW2[u];
    const float pc = silu_f(sc) * cW2[u];

    red[u] = pe;
    __syncthreads();
    for (int s = 32; s > 0; s >>= 1) {
        if (u < s) red[u] += red[u + s];
        __syncthreads();
    }
    if (u == 0) E_row[i] = red[0] + eb2[0];
    __syncthreads();

    red[u] = pc;
    __syncthreads();
    for (int s = 32; s > 0; s >>= 1) {
        if (u < s) red[u] += red[u + s];
        __syncthreads();
    }
    if (u == 0) q_raw[i] = red[0] + cb2[0];
}

// ---------------------------------------------------------------------------
// K5: deterministic fixed-order reduction of E_row and q_raw (single block).
// accums[0] = sum E_row, accums[1] = sum q_raw
// ---------------------------------------------------------------------------
__global__ void reduce_rows_kernel(const float* __restrict__ E_row,
                                   const float* __restrict__ q_raw,
                                   float* __restrict__ accums, int n) {
    __shared__ float s1[256];
    __shared__ float s2[256];
    float a = 0.0f, b = 0.0f;
    for (int i = threadIdx.x; i < n; i += 256) { a += E_row[i]; b += q_raw[i]; }
    s1[threadIdx.x] = a;
    s2[threadIdx.x] = b;
    __syncthreads();
    for (int s = 128; s > 0; s >>= 1) {
        if (threadIdx.x < s) {
            s1[threadIdx.x] += s1[threadIdx.x + s];
            s2[threadIdx.x] += s2[threadIdx.x + s];
        }
        __syncthreads();
    }
    if (threadIdx.x == 0) { accums[0] = s1[0]; accums[1] = s2[0]; }
}

// ---------------------------------------------------------------------------
// K6: conserved charges  q = q_raw + (Q - sum q_raw)/n ; also publish to out.
// ---------------------------------------------------------------------------
__global__ void qfix_kernel(const float* __restrict__ q_raw,
                            const float* __restrict__ accums,
                            const int* __restrict__ Q,
                            float* __restrict__ q_out,   // d_out + 1
                            float* __restrict__ q_buf, int n) {
    int i = blockIdx.x * blockDim.x + threadIdx.x;
    if (i >= n) return;
    const float corr = ((float)Q[0] - accums[1]) / (float)n;
    const float q = q_raw[i] + corr;
    q_buf[i] = q;
    q_out[i] = q;
}

// ---------------------------------------------------------------------------
// K7: shielded Coulomb, one row per block, j > i strided over 256 threads.
// Pair terms recomputed in registers (no N^2 storage). Raw v_rsq/v_log/v_exp:
// shield ~ N(0, 0.05^2) so softplus is well-conditioned at 1-ulp hw accuracy.
// ---------------------------------------------------------------------------
__global__ void coulomb_kernel(const float* __restrict__ Rp,
                               const int* __restrict__ Z,
                               const float* __restrict__ q,
                               const float* __restrict__ shield,
                               float* __restrict__ Ecoul_row, int n) {
    const int i = blockIdx.x;
    const float xi = Rp[3 * i + 0];
    const float yi = Rp[3 * i + 1];
    const float zi = Rp[3 * i + 2];
    const float qi = q[i];
    const int   zi_row = Z[i] * N_ELEM;

    float local = 0.0f;
    for (int j = i + 1 + (int)threadIdx.x; j < n; j += 256) {
        const float dx = xi - Rp[3 * j + 0];
        const float dy = yi - Rp[3 * j + 1];
        const float dz = zi - Rp[3 * j + 2];
        const float r2 = dx * dx + dy * dy + dz * dz;
        const float s  = shield[zi_row + Z[j]];
        const float a  = __logf(1.0f + __expf(s));           // softplus
        local += qi * q[j] * __builtin_amdgcn_rsqf(r2 + a * a);
    }

    __shared__ float red[256];
    red[threadIdx.x] = local;
    __syncthreads();
    for (int s = 128; s > 0; s >>= 1) {
        if (threadIdx.x < s) red[threadIdx.x] += red[threadIdx.x + s];
        __syncthreads();
    }
    if (threadIdx.x == 0) Ecoul_row[i] = red[0];
}

// ---------------------------------------------------------------------------
// K8: finalize: out[0] = E_local + K_E * sum Ecoul_row  (fixed-order sum).
// ---------------------------------------------------------------------------
__global__ void finalize_kernel(const float* __restrict__ Ecoul_row,
                                const float* __restrict__ accums,
                                float* __restrict__ out, int n) {
    __shared__ float s1[256];
    float a = 0.0f;
    for (int i = threadIdx.x; i < n; i += 256) a += Ecoul_row[i];
    s1[threadIdx.x] = a;
    __syncthreads();
    for (int s = 128; s > 0; s >>= 1) {
        if (threadIdx.x < s) s1[threadIdx.x] += s1[threadIdx.x + s];
        __syncthreads();
    }
    if (threadIdx.x == 0) out[0] = accums[0] + K_E * s1[0];
}

// ---------------------------------------------------------------------------
extern "C" void kernel_launch(void* const* d_in, const int* in_sizes, int n_in,
                              void* d_out, int out_size, void* d_ws, size_t ws_size,
                              hipStream_t stream) {
    const int*   Z          = (const int*)  d_in[0];
    const float* R          = (const float*)d_in[1];
    const int*   Q          = (const int*)  d_in[2];
    const int*   S          = (const int*)  d_in[3];
    const float* atom_embed = (const float*)d_in[4];
    const float* q_embed    = (const float*)d_in[5];
    const float* s_embed    = (const float*)d_in[6];
    const float* shield     = (const float*)d_in[7];
    const float* eW1        = (const float*)d_in[8];
    const float* eb1        = (const float*)d_in[9];
    const float* eW2        = (const float*)d_in[10];
    const float* eb2        = (const float*)d_in[11];
    const float* cW1        = (const float*)d_in[12];
    const float* cb1        = (const float*)d_in[13];
    const float* cW2        = (const float*)d_in[14];
    const float* cb2        = (const float*)d_in[15];

    const int n = in_sizes[0];            // 4096

    // workspace carve-up (floats)
    float*    ws        = (float*)d_ws;
    float*    h0        = ws;                                  // n*64
    float*    h         = h0 + (size_t)n * HID;                // n*64
    uint32_t* bpack     = (uint32_t*)(h + (size_t)n * HID);    // n*32 dwords
    float*    E_row     = (float*)(bpack + (size_t)n * 32);    // n
    float*    q_raw     = E_row + n;                           // n
    float*    q_buf     = q_raw + n;                           // n
    float*    Ecoul_row = q_buf + n;                           // n
    float*    accums    = Ecoul_row + n;                       // 4

    float* out = (float*)d_out;

    embed_kernel<<<(n * HID + 255) / 256, 256, 0, stream>>>(
        Z, Q, S, atom_embed, q_embed, s_embed, h0, n);

    repack_kernel<<<(n * 32 + 255) / 256, 256, 0, stream>>>(h0, bpack, n);

    msgpass_kernel<<<n / 128, 256, 0, stream>>>(R, h0, bpack, h, n);

    row_mlp_kernel<<<n, HID, 0, stream>>>(
        h, eW1, eb1, eW2, eb2, cW1, cb1, cW2, cb2, E_row, q_raw, n);

    reduce_rows_kernel<<<1, 256, 0, stream>>>(E_row, q_raw, accums, n);

    qfix_kernel<<<(n + 255) / 256, 256, 0, stream>>>(
        q_raw, accums, Q, out + 1, q_buf, n);

    coulomb_kernel<<<n, 256, 0, stream>>>(R, Z, q_buf, shield, Ecoul_row, n);

    finalize_kernel<<<1, 256, 0, stream>>>(Ecoul_row, accums, out, n);
}
